// RuleGuidedVAE_60043642798752
// MI455X (gfx1250) — matrix operand order repaired and checked by reading, not compiled, not run on hardware
//
#include <hip/hip_runtime.h>

typedef __attribute__((ext_vector_type(16))) _Float16 v16h;
typedef __attribute__((ext_vector_type(8)))  float    v8f;
typedef _Float16 hl16;

#define NP   262144          // B*H*W pixels
#define NPC  16777216        // NP*64 elements

union AF { v16h h; unsigned int u[8]; };

__device__ inline void load_a_frag(AF& a0, AF& a1, const unsigned int* row, int g) {
  // A 16x32 f16 layout: lane holds row M, VGPR v -> K = (v<4?0:16) + g*8 + (v&3)*2
  // row points at 64 contiguous halves (32 u32); frag0 = K 0..31, frag1 = K 32..63
#pragma unroll
  for (int v = 0; v < 8; ++v) {
    int k = ((v < 4) ? 0 : 16) + g * 8 + (v & 3) * 2;
    a0.u[v] = row[k >> 1];
    a1.u[v] = row[16 + (k >> 1)];
  }
}

__device__ inline void load_b_frag(AF& b0, AF& b1, const unsigned int* col, int g) {
  // B 32x16 f16 layout: lane holds column N, VGPR v -> K = (v<4?0:16) + g*8 + (v&3)*2
#pragma unroll
  for (int v = 0; v < 8; ++v) {
    int k = ((v < 4) ? 0 : 16) + g * 8 + (v & 3) * 2;
    b0.u[v] = col[k >> 1];
    b1.u[v] = col[16 + (k >> 1)];
  }
}

__device__ inline v8f wmma16(const AF& a, const AF& b, v8f c) {
  return __builtin_amdgcn_wmma_f32_16x16x32_f16(false, a.h, false, b.h, (short)0, c, false, false);
}

// ---------------- tiny utility kernels ----------------
__global__ void k_zero_loss(float* p) { if (threadIdx.x < 4) p[threadIdx.x] = 0.f; }

__global__ void k_cb_pack(const float* cb, hl16* cb16, float* cnorm) {
  int c = blockIdx.x * blockDim.x + threadIdx.x;
  if (c >= 512) return;
  float n = 0.f;
  for (int k = 0; k < 64; ++k) { float v = cb[c * 64 + k]; cb16[c * 64 + k] = (hl16)v; n += v * v; }
  cnorm[c] = n;
}

// pack OIHW f32 conv weights (first 64 input channels) -> wt2[s][co][ci] f16
__global__ void k_wpack(const float* W, int Cin, hl16* wt2) {
  int t = blockIdx.x * blockDim.x + threadIdx.x;
  if (t >= 9 * 64 * 64) return;
  int ci = t & 63, co = (t >> 6) & 63, s = t >> 12;
  wt2[((s * 64 + co) << 6) + ci] = (hl16)W[((size_t)co * Cin + ci) * 9 + s];
}

// Ppat[b][ph][pw][o] = sum over valid taps of broadcast channels (cin 64..191)
__global__ void k_ppat(const float* r, const float* W, float* ppat) {
  int t = blockIdx.x * blockDim.x + threadIdx.x;
  if (t >= 64 * 9 * 64) return;
  int o = t & 63; int q = t >> 6; int pw = q % 3; int ph = (q / 3) % 3; int b = q / 9;
  int kh0 = (ph == 0) ? 1 : 0, kh1 = (ph == 2) ? 1 : 2;
  int kw0 = (pw == 0) ? 1 : 0, kw1 = (pw == 2) ? 1 : 2;
  float s = 0.f;
  for (int c = 0; c < 128; ++c) {
    float rv = r[b * 128 + c];
    const float* wb = W + ((size_t)o * 192 + 64 + c) * 9;
    for (int kh = kh0; kh <= kh1; ++kh)
      for (int kw = kw0; kw <= kw1; ++kw)
        s += rv * wb[kh * 3 + kw];
  }
  ppat[t] = s;
}

// pixel 1x1 conv: x NCHW [B,64,HW] -> y NHWC [NP,64]
__global__ __launch_bounds__(256) void k_conv1x1(const float* x, const float* w,
                                                 const float* bias, float* y) {
  size_t e = (size_t)blockIdx.x * 256 + threadIdx.x;
  int o = (int)(e & 63); size_t p = e >> 6; int b = (int)(p >> 12); int hw = (int)(p & 4095);
  const float* xp = x + (size_t)b * 64 * 4096 + hw;
  float s = bias[o];
  for (int i = 0; i < 64; ++i) s += xp[(size_t)i * 4096] * w[o * 64 + i];
  y[e] = s;
}

// group-norm stats over (b, group of 8 ch) of y NHWC [NP,64]
__global__ __launch_bounds__(256) void k_gn_stats(const float* y, float* stats) {
  __shared__ float s1[256], s2[256];
  int bg = blockIdx.x; int b = bg >> 3; int gg = bg & 7; int tid = threadIdx.x;
  float a = 0.f, q = 0.f;
  for (int i = tid; i < 4096; i += 256) {
    const float* p = y + ((size_t)(b * 4096 + i)) * 64 + gg * 8;
#pragma unroll
    for (int c = 0; c < 8; ++c) { float v = p[c]; a += v; q += v * v; }
  }
  s1[tid] = a; s2[tid] = q; __syncthreads();
  for (int o = 128; o >= 1; o >>= 1) {
    if (tid < o) { s1[tid] += s1[tid + o]; s2[tid] += s2[tid + o]; }
    __syncthreads();
  }
  if (tid == 0) {
    float m = s1[0] * (1.f / 32768.f);
    float var = s2[0] * (1.f / 32768.f) - m * m;
    stats[2 * bg] = m; stats[2 * bg + 1] = rsqrtf(var + 1e-5f);
  }
}

// mode 1: relu ; mode 2: lrelu*0.1  -> f16 NHWC
__global__ __launch_bounds__(256) void k_gn_apply_f16(const float* y, const float* stats,
                                                      hl16* outp, int mode) {
  size_t e = (size_t)blockIdx.x * 256 + threadIdx.x;
  int c = (int)(e & 63); int b = (int)(e >> 18);
  float m = stats[2 * (b * 8 + (c >> 3))];
  float rs = stats[2 * (b * 8 + (c >> 3)) + 1];
  float v = (y[e] - m) * rs;
  if (mode == 1) v = v > 0.f ? v : 0.f;
  else v = (v >= 0.f ? v : 0.2f * v) * 0.1f;
  outp[e] = (hl16)v;
}

// relu -> f32 NCHW (final comp)
__global__ __launch_bounds__(256) void k_gn_apply_nchw(const float* y, const float* stats,
                                                       float* outp) {
  size_t e = (size_t)blockIdx.x * 256 + threadIdx.x;
  int c = (int)(e & 63); size_t p = e >> 6; int b = (int)(p >> 12); int hw = (int)(p & 4095);
  float m = stats[2 * (b * 8 + (c >> 3))];
  float rs = stats[2 * (b * 8 + (c >> 3)) + 1];
  float v = (y[e] - m) * rs;
  outp[((size_t)(b * 64 + c)) * 4096 + hw] = v > 0.f ? v : 0.f;
}

// ---------------- WMMA pixel VQ: argmin over 512 codes, K=64 ----------------
__global__ __launch_bounds__(256) void k_vq_pixel(const hl16* pz16, const hl16* cb16,
                                                  const float* cnorm, int* idxb) {
  __shared__ unsigned int cbu[256 * 32]; // 256 codes x 64 halves (32KB)
  __shared__ float cns[512];
  int tid = threadIdx.x;
  for (int i = tid; i < 512; i += 256) cns[i] = cnorm[i];
  int wave = tid >> 5, lane = tid & 31, g = lane >> 4, l16 = lane & 15;
  int tile = blockIdx.x * 8 + wave; int pb = tile * 16;

  const unsigned int* zrow = (const unsigned int*)pz16 + (size_t)(pb + l16) * 32;
  AF a0, a1;
  load_a_frag(a0, a1, zrow, g);

  v8f zero = {};
  float best[8]; int bidx[8];
#pragma unroll
  for (int v = 0; v < 8; ++v) { best[v] = 3.0e38f; bidx[v] = 0; }

  for (int hb = 0; hb < 2; ++hb) {
    __syncthreads();
    const unsigned int* src = (const unsigned int*)cb16 + (size_t)hb * 256 * 32;
    for (int i = tid; i < 256 * 32; i += 256) cbu[i] = src[i];
    __syncthreads();
    for (int tn = 0; tn < 16; ++tn) {
      int code_l = tn * 16 + l16;
      AF b0, b1;
      load_b_frag(b0, b1, cbu + code_l * 32, g);
      v8f c = zero;
      c = wmma16(a0, b0, c);
      c = wmma16(a1, b1, c);
      int code = hb * 256 + code_l;
      float cn = cns[code];
#pragma unroll
      for (int v = 0; v < 8; ++v) {
        float sc = cn - 2.f * c[v];
        if (sc < best[v]) { best[v] = sc; bidx[v] = code; }
      }
    }
  }
  // reduce across the 16 lanes of each half-group (each lane covers codes == l16 mod 16)
#pragma unroll
  for (int v = 0; v < 8; ++v) {
    for (int off = 8; off >= 1; off >>= 1) {
      float ob = __shfl_xor(best[v], off, 16);
      int   oi = __shfl_xor(bidx[v], off, 16);
      if (ob < best[v] || (ob == best[v] && oi < bidx[v])) { best[v] = ob; bidx[v] = oi; }
    }
  }
  if (l16 == 0) {
#pragma unroll
    for (int v = 0; v < 8; ++v) idxb[pb + v + 8 * g] = bidx[v];
  }
}

// gather q, in-place replace z(f16) with q(f16), accumulate 1x MSE sum
__global__ __launch_bounds__(256) void k_gather_pixel(hl16* act, const int* idxb,
                                                      const float* cb, float* lossacc) {
  __shared__ float red[256];
  int tid = threadIdx.x;
  size_t e = (size_t)blockIdx.x * 256 + tid;
  int p = (int)(e >> 6); int k = (int)(e & 63);
  int c = idxb[p];
  float qv = cb[c * 64 + k];
  float zv = (float)act[e];
  act[e] = (hl16)qv;
  float d = zv - qv;
  red[tid] = d * d; __syncthreads();
  for (int o = 128; o >= 1; o >>= 1) { if (tid < o) red[tid] += red[tid + o]; __syncthreads(); }
  if (tid == 0) atomicAdd(lossacc, red[0]);
}

// ---------------- WMMA 3x3 conv: x f16 NHWC [NP,64] -> y f32 NHWC [NP,64] ----------------
__global__ __launch_bounds__(256) void k_conv3x3(const hl16* xact, const hl16* wt2,
                                                 const float* bias, const float* ppat,
                                                 float* outp) {
  int tid = threadIdx.x, wave = tid >> 5, lane = tid & 31, g = lane >> 4, l16 = lane & 15;
  int tile = blockIdx.x * 8 + wave; int pb = tile * 16;
  int b = pb >> 12; int h = (pb >> 6) & 63; int wb = pb & 63;
  const unsigned int* xu = (const unsigned int*)xact;
  const unsigned int* wu = (const unsigned int*)wt2;
  v8f zero = {};
  v8f acc[4] = {zero, zero, zero, zero};

  for (int s = 0; s < 9; ++s) {
    int dh = s / 3 - 1, dw = s % 3 - 1;
    int hh = h + dh, ww = wb + l16 + dw;
    AF a0, a1;
    if (hh >= 0 && hh < 64 && ww >= 0 && ww < 64) {
      load_a_frag(a0, a1, xu + ((size_t)((b * 64 + hh) * 64 + ww)) * 32, g);
    } else {
#pragma unroll
      for (int v = 0; v < 8; ++v) { a0.u[v] = 0u; a1.u[v] = 0u; }
    }
#pragma unroll
    for (int nt = 0; nt < 4; ++nt) {
      int co = nt * 16 + l16;
      AF b0, b1;
      load_b_frag(b0, b1, wu + ((size_t)(s * 64 + co)) * 32, g);
      acc[nt] = wmma16(a0, b0, acc[nt]);
      acc[nt] = wmma16(a1, b1, acc[nt]);
    }
  }

  int ph = (h == 0) ? 0 : ((h == 63) ? 2 : 1);
#pragma unroll
  for (int v = 0; v < 8; ++v) {
    int M = v + 8 * g; int pm = pb + M; int w = wb + M;
    int pw = (w == 0) ? 0 : ((w == 63) ? 2 : 1);
    const float* pp = ppat ? (ppat + ((size_t)((b * 3 + ph) * 3 + pw)) * 64) : (const float*)0;
#pragma unroll
    for (int nt = 0; nt < 4; ++nt) {
      int co = nt * 16 + l16;
      float val = acc[nt][v] + bias[co];
      if (pp) val += pp[co];
      outp[(size_t)pm * 64 + co] = val;
    }
  }
}

// ---------------- small dense layers ----------------
__global__ void k_gemm(const float* in, const float* W, const float* bias, float* outp,
                       int R, int K, int N) {
  int t = blockIdx.x * blockDim.x + threadIdx.x;
  if (t >= R * N) return;
  int r = t / N, n = t % N;
  float s = bias ? bias[n] : 0.f;
  const float* ip = in + (size_t)r * K;
  for (int k = 0; k < K; ++k) s += ip[k] * W[(size_t)k * N + n];
  outp[t] = s;
}

// mode 0:none 1:relu 2:lrelu*0.1 3:tanh*0.1
__global__ __launch_bounds__(256) void k_ln_act(const float* in, float* outp, int D, int mode) {
  __shared__ float s1[256], s2[256];
  int row = blockIdx.x, tid = threadIdx.x;
  const float* ip = in + (size_t)row * D;
  float a = 0.f, q = 0.f;
  for (int i = tid; i < D; i += 256) { float v = ip[i]; a += v; q += v * v; }
  s1[tid] = a; s2[tid] = q; __syncthreads();
  for (int o = 128; o >= 1; o >>= 1) {
    if (tid < o) { s1[tid] += s1[tid + o]; s2[tid] += s2[tid + o]; }
    __syncthreads();
  }
  float m = s1[0] / (float)D;
  float var = s2[0] / (float)D - m * m;
  float rs = rsqrtf(var + 1e-5f);
  for (int i = tid; i < D; i += 256) {
    float v = (ip[i] - m) * rs;
    if (mode == 1) v = v > 0.f ? v : 0.f;
    else if (mode == 2) v = (v >= 0.f ? v : 0.2f * v) * 0.1f;
    else if (mode == 3) v = tanhf(v) * 0.1f;
    outp[(size_t)row * D + i] = v;
  }
}

__global__ __launch_bounds__(256) void k_vq_row(const float* z, const float* cb, float* qout,
                                                float* lossacc, int D, int K) {
  __shared__ float zsh[128];
  __shared__ float sd[256];
  __shared__ int si[256];
  int row = blockIdx.x, tid = threadIdx.x;
  if (tid < D) zsh[tid] = z[(size_t)row * D + tid];
  __syncthreads();
  float dist = 3.0e38f;
  if (tid < K) {
    float s = 0.f; const float* c = cb + (size_t)tid * D;
    for (int k = 0; k < D; ++k) { float d = zsh[k] - c[k]; s += d * d; }
    dist = s;
  }
  sd[tid] = dist; si[tid] = tid; __syncthreads();
  for (int o = 128; o >= 1; o >>= 1) {
    if (tid < o) {
      float ov = sd[tid + o]; int oi = si[tid + o];
      if (ov < sd[tid] || (ov == sd[tid] && oi < si[tid])) { sd[tid] = ov; si[tid] = oi; }
    }
    __syncthreads();
  }
  int best = si[0]; __syncthreads();
  float part = 0.f;
  if (tid < D) {
    float qv = cb[(size_t)best * D + tid];
    qout[(size_t)row * D + tid] = qv;
    float d = zsh[tid] - qv; part = d * d;
  }
  sd[tid] = part; __syncthreads();
  for (int o = 128; o >= 1; o >>= 1) { if (tid < o) sd[tid] += sd[tid + o]; __syncthreads(); }
  if (tid == 0) atomicAdd(lossacc, sd[0]);
}

__global__ void k_mean_n(const float* in, float* outp, int B, int N, int C) {
  int t = blockIdx.x * blockDim.x + threadIdx.x;
  if (t >= B * C) return;
  int b = t / C, c = t % C;
  float s = 0.f;
  for (int n = 0; n < N; ++n) s += in[((size_t)b * N + n) * C + c];
  outp[t] = s / (float)N;
}

__global__ void k_concat2(const float* i1, const float* i2, float* outp, int R, int A, int Bc) {
  int t = blockIdx.x * blockDim.x + threadIdx.x;
  int W = A + Bc;
  if (t >= R * W) return;
  int r = t / W, j = t % W;
  outp[t] = (j < A) ? i1[(size_t)r * A + j] : i2[(size_t)r * Bc + (j - A)];
}

__global__ void k_build_tin(const float* oq, const float* rq, float* outp) {
  int t = blockIdx.x * blockDim.x + threadIdx.x;
  if (t >= 1024 * 256) return;
  int r = t >> 8, j = t & 255, b = r >> 4;
  outp[t] = (j < 128) ? oq[r * 128 + j] : rq[b * 128 + (j - 128)];
}

__global__ void k_loss_final(const float* loss, float* outp) {
  outp[NPC] = 1.1f * (loss[0] / 16777216.f + loss[1] / 131072.f + loss[2] / 8192.f);
}

// ---------------- launcher ----------------
extern "C" void kernel_launch(void* const* d_in, const int* in_sizes, int n_in,
                              void* d_out, int out_size, void* d_ws, size_t ws_size,
                              hipStream_t stream) {
  (void)in_sizes; (void)n_in; (void)out_size; (void)ws_size;
  const float* pixf = (const float*)d_in[0];
  const float* objf = (const float*)d_in[1];
  const float* hin  = (const float*)d_in[2];
  const float* hout = (const float*)d_in[3];
  const float* w_pp = (const float*)d_in[4];
  const float* b_pp = (const float*)d_in[5];
  const float* cbp  = (const float*)d_in[6];
  const float* w_op = (const float*)d_in[7];
  const float* b_op = (const float*)d_in[8];
  const float* cbo  = (const float*)d_in[9];
  const float* w_e1 = (const float*)d_in[10];
  const float* b_e1 = (const float*)d_in[11];
  const float* w_e2 = (const float*)d_in[12];
  const float* b_e2 = (const float*)d_in[13];
  const float* w_od = (const float*)d_in[14];
  const float* b_od = (const float*)d_in[15];
  const float* w_rg = (const float*)d_in[16];
  const float* b_rg = (const float*)d_in[17];
  const float* cbr  = (const float*)d_in[18];
  const float* w_t1 = (const float*)d_in[19];
  const float* b_t1 = (const float*)d_in[20];
  const float* w_t2 = (const float*)d_in[21];
  const float* b_t2 = (const float*)d_in[22];
  const float* w_c1 = (const float*)d_in[23];
  const float* b_c1 = (const float*)d_in[24];
  const float* w_c2 = (const float*)d_in[25];
  const float* b_c2 = (const float*)d_in[26];
  const float* w_cc = (const float*)d_in[27];
  const float* b_cc = (const float*)d_in[28];
  float* outp = (float*)d_out;

  char* ws = (char*)d_ws; size_t off = 0;
  auto alloc = [&](size_t bytes) -> void* {
    void* p = ws + off; off = (off + bytes + 255) & ~(size_t)255; return p;
  };
  float* loss   = (float*)alloc(16);
  float* Y      = (float*)alloc((size_t)NPC * 4);        // f32 NHWC scratch (y_pp / y1 / y2 / y3)
  hl16*  ACT    = (hl16*)alloc((size_t)NPC * 2);         // f16 NHWC (pz16 / x1 / x2 / x3)
  int*   idxb   = (int*)alloc((size_t)NP * 4);
  hl16*  cb16   = (hl16*)alloc(512 * 64 * 2);
  float* cnorm  = (float*)alloc(512 * 4);
  hl16*  wt2_1  = (hl16*)alloc(9 * 64 * 64 * 2);
  hl16*  wt2_2  = (hl16*)alloc(9 * 64 * 64 * 2);
  hl16*  wt2_3  = (hl16*)alloc(9 * 64 * 64 * 2);
  float* stats  = (float*)alloc(64 * 8 * 2 * 4);
  float* ppat1  = (float*)alloc(64 * 9 * 64 * 4);
  float* ppat3  = (float*)alloc(64 * 9 * 64 * 4);
  float* pre    = (float*)alloc(1024 * 256 * 4);
  float* o_s    = (float*)alloc(1024 * 128 * 4);
  float* oq     = (float*)alloc(1024 * 128 * 4);
  float* hbuf   = (float*)alloc(64 * 1024 * 4);
  float* h1     = (float*)alloc(64 * 256 * 4);
  float* h2     = (float*)alloc(64 * 256 * 4);
  float* objd   = (float*)alloc(64 * 128 * 4);
  float* od     = (float*)alloc(64 * 64 * 4);
  float* hc     = (float*)alloc(64 * 320 * 4);
  float* rule   = (float*)alloc(64 * 128 * 4);
  float* rq     = (float*)alloc(64 * 128 * 4);
  float* t1     = (float*)alloc(1024 * 256 * 4);
  float* tobj   = (float*)alloc(1024 * 128 * 4);
  float* omean  = (float*)alloc(64 * 128 * 4);

  k_zero_loss<<<1, 32, 0, stream>>>(loss);
  k_cb_pack<<<2, 256, 0, stream>>>(cbp, cb16, cnorm);
  k_wpack<<<144, 256, 0, stream>>>(w_c1, 192, wt2_1);
  k_wpack<<<144, 256, 0, stream>>>(w_c2, 64, wt2_2);
  k_wpack<<<144, 256, 0, stream>>>(w_cc, 192, wt2_3);

  // pixel projector -> GN -> lrelu*0.1 (f16 NHWC) -> WMMA VQ -> gather (x1 = q)
  k_conv1x1<<<65536, 256, 0, stream>>>(pixf, w_pp, b_pp, Y);
  k_gn_stats<<<512, 256, 0, stream>>>(Y, stats);
  k_gn_apply_f16<<<65536, 256, 0, stream>>>(Y, stats, ACT, 2);
  k_vq_pixel<<<2048, 256, 0, stream>>>(ACT, cb16, cnorm, idxb);
  k_gather_pixel<<<65536, 256, 0, stream>>>(ACT, idxb, cbp, loss + 0);

  // object path
  k_gemm<<<512, 256, 0, stream>>>(objf, w_op, b_op, pre, 1024, 128, 128);
  k_ln_act<<<1024, 256, 0, stream>>>(pre, o_s, 128, 2);
  k_vq_row<<<1024, 256, 0, stream>>>(o_s, cbo, oq, loss + 1, 128, 256);

  // rule path
  k_concat2<<<256, 256, 0, stream>>>(hin, hout, hbuf, 64, 512, 512);
  k_gemm<<<64, 256, 0, stream>>>(hbuf, w_e1, b_e1, pre, 64, 1024, 256);
  k_ln_act<<<64, 256, 0, stream>>>(pre, h1, 256, 1);
  k_gemm<<<64, 256, 0, stream>>>(h1, w_e2, b_e2, pre, 64, 256, 256);
  k_ln_act<<<64, 256, 0, stream>>>(pre, h2, 256, 1);
  k_mean_n<<<32, 256, 0, stream>>>(objf, objd, 64, 16, 128);
  k_gemm<<<16, 256, 0, stream>>>(objd, w_od, b_od, pre, 64, 128, 64);
  k_ln_act<<<64, 256, 0, stream>>>(pre, od, 64, 1);
  k_concat2<<<80, 256, 0, stream>>>(h2, od, hc, 64, 256, 64);
  k_gemm<<<32, 256, 0, stream>>>(hc, w_rg, b_rg, pre, 64, 320, 128);
  k_ln_act<<<64, 256, 0, stream>>>(pre, rule, 128, 3);
  k_vq_row<<<64, 256, 0, stream>>>(rule, cbr, rq, loss + 2, 128, 128);

  // transform network + object mean map
  k_build_tin<<<1024, 256, 0, stream>>>(oq, rq, t1);           // t1 used as t_in temp
  k_gemm<<<1024, 256, 0, stream>>>(t1, w_t1, b_t1, pre, 1024, 256, 256);
  k_ln_act<<<1024, 256, 0, stream>>>(pre, t1, 256, 1);
  k_gemm<<<512, 256, 0, stream>>>(t1, w_t2, b_t2, pre, 1024, 256, 128);
  k_ln_act<<<1024, 256, 0, stream>>>(pre, tobj, 128, 0);
  k_mean_n<<<32, 256, 0, stream>>>(tobj, omean, 64, 16, 128);

  // broadcast-channel conv contributions (rank-1 per border pattern)
  k_ppat<<<144, 256, 0, stream>>>(rq, w_c1, ppat1);
  k_ppat<<<144, 256, 0, stream>>>(omean, w_cc, ppat3);

  // conv1 (pq + rule term) -> GN relu -> conv2 -> GN relu -> composer conv -> GN relu (NCHW)
  k_conv3x3<<<2048, 256, 0, stream>>>(ACT, wt2_1, b_c1, ppat1, Y);
  k_gn_stats<<<512, 256, 0, stream>>>(Y, stats);
  k_gn_apply_f16<<<65536, 256, 0, stream>>>(Y, stats, ACT, 1);
  k_conv3x3<<<2048, 256, 0, stream>>>(ACT, wt2_2, b_c2, (const float*)0, Y);
  k_gn_stats<<<512, 256, 0, stream>>>(Y, stats);
  k_gn_apply_f16<<<65536, 256, 0, stream>>>(Y, stats, ACT, 1);
  k_conv3x3<<<2048, 256, 0, stream>>>(ACT, wt2_3, b_cc, ppat3, Y);
  k_gn_stats<<<512, 256, 0, stream>>>(Y, stats);
  k_gn_apply_nchw<<<65536, 256, 0, stream>>>(Y, stats, outp);

  k_loss_final<<<1, 1, 0, stream>>>(loss, outp);
}
